// SwinTransformerBlock_8108898255019
// MI455X (gfx1250) — matrix operand order repaired and checked by reading, not compiled, not run on hardware
//
#include <hip/hip_runtime.h>
#include <math.h>

// ---------------------------------------------------------------------------
// Types for CDNA5 WMMA (wave32): bf16 inputs, f32 accumulate.
// ---------------------------------------------------------------------------
typedef __bf16 bf16_t;
typedef bf16_t v16bf __attribute__((ext_vector_type(16)));
typedef bf16_t v8bf  __attribute__((ext_vector_type(8)));
typedef float  v8f   __attribute__((ext_vector_type(8)));

union Frag16 { v16bf v; v8bf h[2]; };

#define WMMA_BF16(A, B, C) \
  __builtin_amdgcn_wmma_f32_16x16x32_bf16(false, (A), false, (B), (short)0, (C), false, false)

// Problem constants
// B=2, T=64, H=32, W=32, C=192, NH=6, HD=32, WT=8, WE=(4,4), ST=4, SE=(2,2)
// N=128 tokens/window, NW=512 windows/image, total windows = 1024, tokens = 131072

// window-row (win*128+n) -> global token index (applies cyclic shift +4,+2,+2)
__device__ __forceinline__ int win_row_to_global(int row) {
  int w  = row >> 7;           // window id 0..1023
  int n  = row & 127;
  int b  = w >> 9;             // image 0..1
  int wr = w & 511;
  int wt = wr >> 6;            // 0..7
  int wh = (wr >> 3) & 7;
  int ww = wr & 7;
  int tn = n >> 4;             // 0..7
  int hn = (n >> 2) & 3;
  int wn = n & 3;
  int t = (wt * 8 + tn + 4) & 63;
  int h = (wh * 4 + hn + 2) & 31;
  int wc = (ww * 4 + wn + 2) & 31;
  return ((b * 64 + t) * 32 + h) * 32 + wc;
}

// shift-mask region id of a token (unshifted image coordinates)
__device__ __forceinline__ int region3(int wt, int wh, int ww, int tn, int hn, int wn) {
  int t = wt * 8 + tn;
  int h = wh * 4 + hn;
  int w = ww * 4 + wn;
  int tp = (t < 56) ? 0 : ((t < 60) ? 1 : 2);
  int hp = (h < 28) ? 0 : ((h < 30) ? 1 : 2);
  int wp = (w < 28) ? 0 : ((w < 30) ? 1 : 2);
  return tp * 9 + hp * 3 + wp;
}

// ---------------------------------------------------------------------------
// fp32 -> bf16 convert (weights)
// ---------------------------------------------------------------------------
__global__ __launch_bounds__(256) void f2bf_kernel(const float* __restrict__ in,
                                                   bf16_t* __restrict__ out, int n) {
  int i = blockIdx.x * 256 + threadIdx.x;
  if (i < n) out[i] = (bf16_t)in[i];
}

// ---------------------------------------------------------------------------
// LayerNorm over C=192, one wave per token row.
// mapped=1: output row is window-row, source token is shifted/partitioned.
// mapped=0: plain token rows (LN2).
// ---------------------------------------------------------------------------
__global__ __launch_bounds__(256) void ln_kernel(const float* __restrict__ x,
                                                 const float* __restrict__ g,
                                                 const float* __restrict__ b,
                                                 bf16_t* __restrict__ out, int mapped) {
  int row  = blockIdx.x * 8 + (threadIdx.x >> 5);
  int lane = threadIdx.x & 31;
  int src  = mapped ? win_row_to_global(row) : row;
  const float* xr = x + (size_t)src * 192;
  float v[6];
  float s = 0.f, ss = 0.f;
  for (int i = 0; i < 6; ++i) {
    v[i] = xr[lane + i * 32];
    s += v[i];
    ss += v[i] * v[i];
  }
  for (int o = 16; o > 0; o >>= 1) {
    s  += __shfl_xor(s, o);
    ss += __shfl_xor(ss, o);
  }
  float mu   = s * (1.f / 192.f);
  float var  = ss * (1.f / 192.f) - mu * mu;
  float rstd = rsqrtf(var + 1e-5f);
  bf16_t* orow = out + (size_t)row * 192;
  for (int i = 0; i < 6; ++i) {
    int c = lane + i * 32;
    orow[c] = (bf16_t)((v[i] - mu) * rstd * g[c] + b[c]);
  }
}

// ---------------------------------------------------------------------------
// Generic WMMA GEMM: D[M,N] = A[M,K] @ W[K,N] + bias, bf16 in, f32 acc.
// Block tile 128x64, 256 threads = 8 waves, each wave owns a 16x64 strip.
// A tile is staged with CDNA5 async global->LDS copies (ASYNCcnt).
// Modes: 1=QKV scatter (+scale q), 2=GELU->bf16, 3=proj residual scatter,
//        4=final residual -> d_out
// ---------------------------------------------------------------------------
__global__ __launch_bounds__(256) void gemm_bf16_kernel(
    const bf16_t* __restrict__ A, const bf16_t* __restrict__ W,
    const float* __restrict__ bias, int K, int lda, int ldw, int mode,
    bf16_t* out_bf, int ldo, bf16_t* qb, bf16_t* kb, bf16_t* vb,
    const float* resid_in, float* out_f) {
  __shared__ bf16_t At[128 * 40];
  __shared__ bf16_t Wt[64 * 40];

  const int tid   = threadIdx.x;
  const int wv    = tid >> 5;
  const int lane  = tid & 31;
  const int half  = lane >> 4;
  const int lm    = lane & 15;
  const int mBase = blockIdx.x * 128;
  const int nBase = blockIdx.y * 64;

  v8f acc[4];
  for (int i = 0; i < 4; ++i)
    for (int e = 0; e < 8; ++e) acc[i][e] = 0.f;

  // per-thread A-tile chunk coordinates (2 x 16B chunks per thread)
  const int c0 = tid * 2;

  for (int k0 = 0; k0 < K; k0 += 32) {
    __syncthreads();
    // ---- A tile 128x32 via async global->LDS (CDNA5 path, ASYNCcnt) ----
    for (int i = 0; i < 2; ++i) {
      int c = c0 + i;
      int r = c >> 2, seg = c & 3;
      unsigned ldsb = (unsigned)(size_t)(&At[r * 40 + seg * 8]);
      unsigned goff = (unsigned)(((size_t)(mBase + r) * lda + k0 + seg * 8) * sizeof(bf16_t));
      asm volatile("global_load_async_to_lds_b128 %0, %1, %2"
                   :: "v"(ldsb), "v"(goff), "s"(A)
                   : "memory");
    }
    // ---- W tile 32x64 -> transposed Wt[n][k] (VGPR round-trip) ----
    {
      int kk = tid >> 3, seg = tid & 7;
      const bf16_t* wp = &W[(size_t)(k0 + kk) * ldw + nBase + seg * 8];
      v8bf w = *(const v8bf*)wp;
      // prefetch next K-step's weight row (non-blocking speculative prefetch)
      if (k0 + 32 < K) __builtin_prefetch(wp + (size_t)32 * ldw, 0, 1);
      for (int e = 0; e < 8; ++e) Wt[(seg * 8 + e) * 40 + kk] = w[e];
    }
    asm volatile("s_wait_asynccnt 0" ::: "memory");
    __syncthreads();

    Frag16 a;
    a.h[0] = *(const v8bf*)&At[(wv * 16 + lm) * 40 + half * 8];
    a.h[1] = *(const v8bf*)&At[(wv * 16 + lm) * 40 + 16 + half * 8];
    for (int nt = 0; nt < 4; ++nt) {
      Frag16 b;
      b.h[0] = *(const v8bf*)&Wt[(nt * 16 + lm) * 40 + half * 8];
      b.h[1] = *(const v8bf*)&Wt[(nt * 16 + lm) * 40 + 16 + half * 8];
      acc[nt] = WMMA_BF16(a.v, b.v, acc[nt]);
    }
  }

  for (int nt = 0; nt < 4; ++nt) {
    for (int r = 0; r < 8; ++r) {
      int mrow = mBase + wv * 16 + half * 8 + r;
      int ncol = nBase + nt * 16 + lm;
      float val = acc[nt][r] + bias[ncol];
      if (mode == 1) {
        int which = ncol / 192;
        int cc = ncol - which * 192;
        int hd = cc >> 5, d = cc & 31;
        size_t win = (size_t)(mrow >> 7);
        int n = mrow & 127;
        size_t dst = ((win * 6 + hd) * 128 + (size_t)n) * 32 + d;
        if (which == 0)
          qb[dst] = (bf16_t)(val * 0.17677669529663687f);  // 1/sqrt(32)
        else if (which == 1)
          kb[dst] = (bf16_t)val;
        else
          vb[dst] = (bf16_t)val;
      } else if (mode == 2) {
        float gv = 0.5f * val * (1.0f + erff(val * 0.7071067811865475f));
        out_bf[(size_t)mrow * ldo + ncol] = (bf16_t)gv;
      } else if (mode == 3) {
        int gidx = win_row_to_global(mrow);
        size_t o = (size_t)gidx * 192 + ncol;
        out_f[o] = resid_in[o] + val;
      } else {  // mode 4
        size_t o = (size_t)mrow * 192 + ncol;
        out_f[o] = resid_in[o] + val;
      }
    }
  }
}

// ---------------------------------------------------------------------------
// Windowed attention, one block per (window, head). S kept in LDS (f32),
// softmax per row, P (bf16) @ V via WMMA with V^T staged in LDS.
// ---------------------------------------------------------------------------
__global__ __launch_bounds__(256) void attn_kernel(
    const bf16_t* __restrict__ q, const bf16_t* __restrict__ k,
    const bf16_t* __restrict__ v, const float* __restrict__ rpb,
    bf16_t* __restrict__ out) {
  __shared__ float  S[128 * 136];
  __shared__ bf16_t P[128 * 136];
  __shared__ bf16_t Vt[32 * 136];

  const int bid  = blockIdx.x;
  const int win  = bid / 6;
  const int head = bid % 6;
  const size_t base = (size_t)bid * 128 * 32;
  const bf16_t* qh = q + base;
  const bf16_t* kh = k + base;
  const bf16_t* vh = v + base;

  const int tid  = threadIdx.x;
  const int wv   = tid >> 5;
  const int lane = tid & 31;
  const int half = lane >> 4;
  const int lm   = lane & 15;

  const int wimg = win & 511;
  const int wt = wimg >> 6, wh = (wimg >> 3) & 7, ww = wimg & 7;

  // stage V^T[d][n]
  {
    int n = tid >> 1;
    int dbase = (tid & 1) * 16;
    v8bf v0 = *(const v8bf*)(vh + n * 32 + dbase);
    v8bf v1 = *(const v8bf*)(vh + n * 32 + dbase + 8);
    for (int e = 0; e < 8; ++e) {
      Vt[(dbase + e) * 136 + n]     = v0[e];
      Vt[(dbase + 8 + e) * 136 + n] = v1[e];
    }
  }

  // ---- S = (q*scale) @ k^T, +rel-pos-bias +shift-mask ----
  Frag16 a;
  {
    const bf16_t* ar = qh + (wv * 16 + lm) * 32;
    a.h[0] = *(const v8bf*)(ar + half * 8);
    a.h[1] = *(const v8bf*)(ar + 16 + half * 8);
  }
  for (int ct = 0; ct < 8; ++ct) {
    Frag16 b;
    const bf16_t* br = kh + (ct * 16 + lm) * 32 + half * 8;
    b.h[0] = *(const v8bf*)(br);
    b.h[1] = *(const v8bf*)(br + 16);
    v8f c;
    for (int e = 0; e < 8; ++e) c[e] = 0.f;
    c = WMMA_BF16(a.v, b.v, c);
    for (int r = 0; r < 8; ++r) {
      int n_ = wv * 16 + half * 8 + r;
      int m_ = ct * 16 + lm;
      int tn = n_ >> 4, hn = (n_ >> 2) & 3, wn = n_ & 3;
      int tm = m_ >> 4, hm = (m_ >> 2) & 3, wm = m_ & 3;
      int ridx = (tn - tm + 7) * 15 + (hn - hm + 3) * 7 + (wn - wm + 3);
      float bias = rpb[ridx * 6 + head];
      int rn = region3(wt, wh, ww, tn, hn, wn);
      int rm = region3(wt, wh, ww, tm, hm, wm);
      float mv = (rn == rm) ? 0.f : -100.f;
      S[n_ * 136 + m_] = c[r] + bias + mv;
    }
  }
  __syncthreads();

  // ---- softmax rows -> P (bf16) ----
  if (tid < 128) {
    float* row = S + tid * 136;
    float mx = -3.4e38f;
    for (int j = 0; j < 128; ++j) mx = fmaxf(mx, row[j]);
    float sum = 0.f;
    for (int j = 0; j < 128; ++j) {
      float e = expf(row[j] - mx);
      row[j] = e;
      sum += e;
    }
    float inv = 1.f / sum;
    bf16_t* pr = P + tid * 136;
    for (int j = 0; j < 128; ++j) pr[j] = (bf16_t)(row[j] * inv);
  }
  __syncthreads();

  // ---- O = P @ V ----
  v8f acc[2];
  for (int i = 0; i < 2; ++i)
    for (int e = 0; e < 8; ++e) acc[i][e] = 0.f;
  for (int ks = 0; ks < 4; ++ks) {
    Frag16 pa;
    const bf16_t* pr = P + (wv * 16 + lm) * 136 + ks * 32;
    pa.h[0] = *(const v8bf*)(pr + half * 8);
    pa.h[1] = *(const v8bf*)(pr + 16 + half * 8);
    for (int ct = 0; ct < 2; ++ct) {
      Frag16 b;
      const bf16_t* vr = Vt + (ct * 16 + lm) * 136 + ks * 32 + half * 8;
      b.h[0] = *(const v8bf*)(vr);
      b.h[1] = *(const v8bf*)(vr + 16);
      acc[ct] = WMMA_BF16(pa.v, b.v, acc[ct]);
    }
  }
  for (int ct = 0; ct < 2; ++ct) {
    for (int r = 0; r < 8; ++r) {
      int n_ = wv * 16 + half * 8 + r;
      int d_ = ct * 16 + lm;
      out[((size_t)win * 128 + n_) * 192 + head * 32 + d_] = (bf16_t)acc[ct][r];
    }
  }
}

// ---------------------------------------------------------------------------
// Host launcher
// ---------------------------------------------------------------------------
extern "C" void kernel_launch(void* const* d_in, const int* in_sizes, int n_in,
                              void* d_out, int out_size, void* d_ws, size_t ws_size,
                              hipStream_t stream) {
  (void)in_sizes; (void)n_in; (void)out_size; (void)ws_size;
  const float* x      = (const float*)d_in[0];
  const float* n1g    = (const float*)d_in[1];
  const float* n1b    = (const float*)d_in[2];
  const float* qkv_w  = (const float*)d_in[3];
  const float* qkv_b  = (const float*)d_in[4];
  const float* rpb    = (const float*)d_in[5];
  const float* proj_w = (const float*)d_in[6];
  const float* proj_b = (const float*)d_in[7];
  const float* n2g    = (const float*)d_in[8];
  const float* n2b    = (const float*)d_in[9];
  const float* fc1_w  = (const float*)d_in[10];
  const float* fc1_b  = (const float*)d_in[11];
  const float* fc2_w  = (const float*)d_in[12];
  const float* fc2_b  = (const float*)d_in[13];
  float* out = (float*)d_out;

  char* ws = (char*)d_ws;
  // workspace layout (bytes)
  bf16_t* wqkv   = (bf16_t*)(ws + 0);            // 192*576
  bf16_t* wproj  = (bf16_t*)(ws + 221184);       // 192*192
  bf16_t* wfc1   = (bf16_t*)(ws + 294912);       // 192*768
  bf16_t* wfc2   = (bf16_t*)(ws + 589824);       // 768*192
  bf16_t* hwin   = (bf16_t*)(ws + (1ull << 20)); // 131072*192  (reused as x2bf)
  bf16_t* qb     = (bf16_t*)(ws + 51380224ull);  // 1024*6*128*32 each
  bf16_t* kb     = qb + 25165824ull;
  bf16_t* vb     = kb + 25165824ull;
  bf16_t* attn_o = (bf16_t*)(ws + 202375168ull); // 131072*192
  bf16_t* hid    = qb;                           // reuse qkv+attn_o: 131072*768
  bf16_t* x2bf   = hwin;                         // reuse
  float*  x2     = (float*)(ws + 252706816ull);  // 131072*192 f32

  // convert weights to bf16
  f2bf_kernel<<<(110592 + 255) / 256, 256, 0, stream>>>(qkv_w, wqkv, 110592);
  f2bf_kernel<<<(36864 + 255) / 256, 256, 0, stream>>>(proj_w, wproj, 36864);
  f2bf_kernel<<<(147456 + 255) / 256, 256, 0, stream>>>(fc1_w, wfc1, 147456);
  f2bf_kernel<<<(147456 + 255) / 256, 256, 0, stream>>>(fc2_w, wfc2, 147456);

  // LN1 + shift + window partition
  ln_kernel<<<16384, 256, 0, stream>>>(x, n1g, n1b, hwin, 1);

  // QKV GEMM: (131072x192)@(192x576) -> scatter q(scaled)/k/v
  gemm_bf16_kernel<<<dim3(1024, 9), 256, 0, stream>>>(
      hwin, wqkv, qkv_b, 192, 192, 576, 1, nullptr, 0, qb, kb, vb, nullptr, nullptr);

  // windowed attention (1024 windows x 6 heads)
  attn_kernel<<<6144, 256, 0, stream>>>(qb, kb, vb, rpb, attn_o);

  // proj GEMM + window-reverse + unshift + residual -> x2 (f32)
  gemm_bf16_kernel<<<dim3(1024, 3), 256, 0, stream>>>(
      attn_o, wproj, proj_b, 192, 192, 192, 3, nullptr, 0, nullptr, nullptr, nullptr, x, x2);

  // LN2 -> bf16
  ln_kernel<<<16384, 256, 0, stream>>>(x2, n2g, n2b, x2bf, 0);

  // fc1 + exact GELU -> hidden (bf16)
  gemm_bf16_kernel<<<dim3(1024, 12), 256, 0, stream>>>(
      x2bf, wfc1, fc1_b, 192, 192, 768, 2, hid, 768, nullptr, nullptr, nullptr, nullptr, nullptr);

  // fc2 + residual -> d_out (f32)
  gemm_bf16_kernel<<<dim3(1024, 3), 256, 0, stream>>>(
      hid, wfc2, fc2_b, 768, 768, 192, 4, nullptr, 0, nullptr, nullptr, nullptr, x2, out);
}
